// attention_84464826843938
// MI455X (gfx1250) — matrix operand order repaired and checked by reading, not compiled, not run on hardware
//
#include <hip/hip_runtime.h>

#define B_   256
#define R_   64
#define F_   2048
#define H_   512
#define DIM_ 512

typedef __attribute__((ext_vector_type(16))) __bf16 v16bf;
typedef __attribute__((ext_vector_type(8)))  float  v8f;

union Tile32 { uint4 u[2]; v16bf v; };

__device__ __forceinline__ unsigned short f32_to_bf16(float f) {
  unsigned int u = __float_as_uint(f);
  u += 0x7FFFu + ((u >> 16) & 1u);   // round-to-nearest-even
  return (unsigned short)(u >> 16);
}

// W_w [F,DIM] fp32 -> wt [DIM,F] bf16, 64x64 LDS tile transpose:
// coalesced read along d, coalesced write along f (avoids 32x line amplification).
__global__ __launch_bounds__(256)
void k_wt_convert(const float* __restrict__ W_w, unsigned short* __restrict__ wt) {
  __shared__ unsigned short tile[64][65];          // +1 pad: transposed reads conflict-free
  const int tid = threadIdx.x;
  const int bf  = blockIdx.x & 31;                 // F_/64 = 32 tiles
  const int bd  = blockIdx.x >> 5;                 // DIM_/64 = 8 tiles
  const int f0  = bf * 64, d0 = bd * 64;
#pragma unroll
  for (int i = 0; i < 16; ++i) {
    int e = tid + i * 256;                         // 0..4095
    int fr = e >> 6, dc = e & 63;
    tile[fr][dc] = f32_to_bf16(W_w[(size_t)(f0 + fr) * DIM_ + d0 + dc]);
  }
  __syncthreads();
#pragma unroll
  for (int i = 0; i < 16; ++i) {
    int e = tid + i * 256;
    int dr = e >> 6, fc = e & 63;
    wt[(size_t)(d0 + dr) * F_ + f0 + fc] = tile[fc][dr];
  }
}

// p[b,d] = W2_b[d] + sum_h prev[b,h]*W2_w[h,d]
// Register-blocked over 8 batches per thread: each W2_w load is reused 8x
// (L2 traffic 268MB -> ~34MB); prev loads are wave-uniform broadcasts.
__global__ __launch_bounds__(256)
void k_prev_proj(const float* __restrict__ prev, const float* __restrict__ W2_w,
                 const float* __restrict__ W2_b, float* __restrict__ pbuf) {
  int idx = blockIdx.x * blockDim.x + threadIdx.x;  // (B_/8)*DIM_ = 16384
  int d  = idx & (DIM_ - 1);
  int b0 = (idx >> 9) * 8;
  float s[8];
  const float bias = W2_b[d];
#pragma unroll
  for (int t = 0; t < 8; ++t) s[t] = bias;
  for (int h = 0; h < H_; ++h) {
    const float w = W2_w[(size_t)h * DIM_ + d];
#pragma unroll
    for (int t = 0; t < 8; ++t)
      s[t] = fmaf(prev[(size_t)(b0 + t) * H_ + h], w, s[t]);
  }
#pragma unroll
  for (int t = 0; t < 8; ++t) pbuf[(size_t)(b0 + t) * DIM_ + d] = s[t];
}

#define LDS_STRIDE 72   // 64 K-elems + 8 bf16 pad => 144B row stride, ds_load_b128 conflict-free
#define NCHUNK (F_ / 64)

__device__ __forceinline__ void stage_load(float (&regs)[16], const float* __restrict__ featB,
                                           int k0, int tid) {
#pragma unroll
  for (int i = 0; i < 16; ++i) {
    int idx = tid + i * 256;
    int r = idx >> 6, c = idx & 63;
    regs[i] = featB[(size_t)r * F_ + k0 + c];
  }
}

__device__ __forceinline__ void stage_store(const float (&regs)[16], unsigned short* buf, int tid) {
#pragma unroll
  for (int i = 0; i < 16; ++i) {
    int idx = tid + i * 256;
    int r = idx >> 6, c = idx & 63;
    buf[r * LDS_STRIDE + c] = f32_to_bf16(regs[i]);
  }
}

__global__ __launch_bounds__(256)
void k_attend(const float* __restrict__ feat, const unsigned short* __restrict__ wt,
              const float* __restrict__ pbuf, const float* __restrict__ W_b,
              const float* __restrict__ W3_w, float* __restrict__ out) {
  __shared__ unsigned short ldsA[2][R_ * LDS_STRIDE];   // double-buffered A stage (2 x 9 KB)
  __shared__ float s_score[R_];
  __shared__ float s_aw[R_];

  const int b     = blockIdx.x;
  const int tid   = threadIdx.x;
  const int wave  = tid >> 5;       // 8 wave32s per block
  const int lane  = tid & 31;
  const int l15   = lane & 15;
  const int hi    = lane >> 4;      // lane half
  const int koffA = hi * 8;         // A: 16-bit 16x32 layout, halves interleave K by 8
  const int koffB = hi * 8;         // B: same K interleave as A (halves interleave K by 8)
  const float* featB = feat + (size_t)b * R_ * F_;

  if (tid < R_) s_score[tid] = 0.0f;

  v8f acc[4][4];                    // [mi][ni] : M=64 x N=64 slab per wave, fp32 accum
#pragma unroll
  for (int mi = 0; mi < 4; ++mi)
#pragma unroll
    for (int ni = 0; ni < 4; ++ni)
      acc[mi][ni] = (v8f)0.0f;

  // prologue: stage chunk 0
  float stg[16];
  stage_load(stg, featB, 0, tid);
  stage_store(stg, &ldsA[0][0], tid);
  __syncthreads();

  for (int c = 0; c < NCHUNK; ++c) {
    const int k0 = c * 64;
    const unsigned short* cur = &ldsA[c & 1][0];

    // issue next chunk's global loads early: latency hides under the 32 WMMAs below
    if (c + 1 < NCHUNK) stage_load(stg, featB, k0 + 64, tid);

#pragma unroll
    for (int kk = 0; kk < 64; kk += 32) {
      Tile32 a[4];
#pragma unroll
      for (int mi = 0; mi < 4; ++mi) {
        const unsigned short* ap = &cur[(mi * 16 + l15) * LDS_STRIDE + kk + koffA];
        a[mi].u[0] = *(const uint4*)(ap);        // K+0..7  (or K+8..15 for hi half)
        a[mi].u[1] = *(const uint4*)(ap + 16);   // K+16..23 (or K+24..31)
      }
#pragma unroll
      for (int ni = 0; ni < 4; ++ni) {
        const int N = wave * 64 + ni * 16 + l15;
        const unsigned short* bp = wt + (size_t)N * F_ + k0 + kk + koffB;
        Tile32 bt;
        bt.u[0] = *(const uint4*)(bp);
        bt.u[1] = *(const uint4*)(bp + 16);
#pragma unroll
        for (int mi = 0; mi < 4; ++mi) {
          acc[mi][ni] = __builtin_amdgcn_wmma_f32_16x16x32_bf16(
              false, a[mi].v, false, bt.v, (short)0, acc[mi][ni], false, false);
        }
      }
    }

    // write next chunk into the other buffer (safe: it was last read before the
    // previous iteration's barrier), then publish for next iteration
    if (c + 1 < NCHUNK) stage_store(stg, &ldsA[(c + 1) & 1][0], tid);
    __syncthreads();
  }

  // score[m] = sum_d tanh(c[m,d]) * W3[d]  -- reduce straight out of accumulators.
  // (W3_b dropped: softmax is shift-invariant.)
  float part[4][8];
#pragma unroll
  for (int mi = 0; mi < 4; ++mi)
#pragma unroll
    for (int j = 0; j < 8; ++j) part[mi][j] = 0.0f;

  const float* pb = pbuf + (size_t)b * DIM_;
#pragma unroll
  for (int ni = 0; ni < 4; ++ni) {
    const int N = wave * 64 + ni * 16 + l15;
    const float bias = W_b[N] + pb[N];
    const float w3   = W3_w[N];
#pragma unroll
    for (int mi = 0; mi < 4; ++mi)
#pragma unroll
      for (int j = 0; j < 8; ++j)
        part[mi][j] = fmaf(tanhf(acc[mi][ni][j] + bias), w3, part[mi][j]);
  }
#pragma unroll
  for (int mi = 0; mi < 4; ++mi)
#pragma unroll
    for (int j = 0; j < 8; ++j)
      atomicAdd(&s_score[mi * 16 + hi * 8 + j], part[mi][j]);   // ds_add_f32
  __syncthreads();

  // softmax over 64 rows
  if (tid < R_) {
    float m = s_score[0];
    for (int r = 1; r < R_; ++r) m = fmaxf(m, s_score[r]);
    float s = 0.0f;
    for (int r = 0; r < R_; ++r) s += expf(s_score[r] - m);
    s_aw[tid] = expf(s_score[tid] - m) / s;
  }
  __syncthreads();

  // q[b,f] = sum_r aw[r] * features[b,r,f]  (fp32, L2-hot re-read, coalesced per lane)
#pragma unroll
  for (int i = 0; i < 8; ++i) {
    const int f = tid + i * 256;
    float q = 0.0f;
    for (int r = 0; r < R_; ++r) q = fmaf(s_aw[r], featB[(size_t)r * F_ + f], q);
    out[(size_t)b * F_ + f] = q;
  }
}

extern "C" void kernel_launch(void* const* d_in, const int* in_sizes, int n_in,
                              void* d_out, int out_size, void* d_ws, size_t ws_size,
                              hipStream_t stream) {
  (void)in_sizes; (void)n_in; (void)out_size; (void)ws_size;
  const float* features = (const float*)d_in[0];
  const float* prev     = (const float*)d_in[1];
  const float* W_w      = (const float*)d_in[2];
  const float* W_b      = (const float*)d_in[3];
  const float* W2_w     = (const float*)d_in[4];
  const float* W2_b     = (const float*)d_in[5];
  const float* W3_w     = (const float*)d_in[6];
  // d_in[7] = W3_b: unused (softmax shift-invariance)
  float* out = (float*)d_out;

  unsigned short* wt   = (unsigned short*)d_ws;                                  // 2 MB
  float*          pbuf = (float*)((char*)d_ws + (size_t)DIM_ * F_ * sizeof(unsigned short)); // 512 KB

  k_wt_convert<<<(DIM_ * F_) / (64 * 64), 256, 0, stream>>>(W_w, wt);
  k_prev_proj <<<(B_ / 8) * DIM_ / 256, 256, 0, stream>>>(prev, W2_w, W2_b, pbuf);
  k_attend    <<<B_, 256, 0, stream>>>(features, wt, pbuf, W_b, W3_w, out);
}
